// GAT_2138893713530
// MI455X (gfx1250) — matrix operand order repaired and checked by reading, not compiled, not run on hardware
//
#include <hip/hip_runtime.h>
#include <hip/hip_bf16.h>

// GAT forward for N=4096, H_STATE=H_LEFT=2 (H=4 after concat), D=256, slope=0.2.
// All matrix math on v_wmma_f32_16x16x32_f16 (f16 inputs, f32 accumulate).
// Softmax is max-stabilized so the f16 probability tile is overflow-safe.
// Attention hot loop uses a ping-pong LDS weight tile: one s_barrier per j-step.

typedef __attribute__((ext_vector_type(16))) _Float16 v16h;
typedef __attribute__((ext_vector_type(8)))  float    v8f;

#define N_TOK 4096
#define HEADS 4
#define DIM   256
#define ROWS  (N_TOK * HEADS)   // 16384
#define MI    32                // i-rows per attention workgroup

__device__ __forceinline__ float lrelu(float x) { return x >= 0.f ? x : 0.2f * x; }

// ---------------------------------------------------------------------------
// dst[m, n] = sum_k A[m,k] * W[n,k] + bias[n]
// mode 0: A row m comes from concat(left, state) (h<2 -> left, h>=2 -> state)
// mode 1: A = xin
// One 16x16 output tile per wave; K-loop = 8 x wmma_f32_16x16x32_f16.
// ---------------------------------------------------------------------------
__global__ __launch_bounds__(256) void lin_kernel(
    int mode,
    const float* __restrict__ left, const float* __restrict__ state,
    const float* __restrict__ xin,
    const float* __restrict__ W,   const float* __restrict__ bias,
    float* __restrict__ dst)
{
    const int lane = threadIdx.x & 31;
    const int wv   = threadIdx.x >> 5;
    const int hf   = lane >> 4;      // lane half
    const int mr   = lane & 15;
    const int m0   = blockIdx.x * 16;
    const int n0   = (blockIdx.y * 8 + wv) * 16;

    const int m = m0 + mr;
    const float* arow;
    if (mode == 0) {
        const int n  = m >> 2;
        const int hh = m & 3;
        arow = (hh < 2) ? (left  + ((size_t)n * 2 + hh)       * DIM)
                        : (state + ((size_t)n * 2 + (hh - 2)) * DIM);
    } else {
        arow = xin + (size_t)m * DIM;
    }
    const float* wrow = W + (size_t)(n0 + mr) * DIM;

    v8f acc = {};
    #pragma unroll
    for (int k0 = 0; k0 < DIM; k0 += 32) {
        // A fragment (16x32 f16): lane-half selects K 0..15 / 16..31
        const int kb0 = k0 + 16 * hf;
        v16h a, b;
        #pragma unroll
        for (int t = 0; t < 8; ++t) {
            a[t]     = (_Float16)arow[kb0 + t];
            a[8 + t] = (_Float16)arow[kb0 + 8 + t];
        }
        // B fragment (32x16 f16): B[k][n] = W[n][k]; lane-half selects K 0..15 / 16..31
        const int kb = k0 + 16 * hf;
        #pragma unroll
        for (int t = 0; t < 16; ++t) b[t] = (_Float16)wrow[kb + t];

        acc = __builtin_amdgcn_wmma_f32_16x16x32_f16(false, a, false, b,
                                                     (short)0, acc, false, false);
    }

    const float bv = bias[n0 + mr];
    #pragma unroll
    for (int r = 0; r < 8; ++r)
        dst[(size_t)(m0 + r + 8 * hf) * DIM + n0 + mr] = acc[r] + bv;
}

// ---------------------------------------------------------------------------
// el[r] = dot(g[r,:], attn_w[0:256]) ; er[r] = dot(g[r,:], attn_w[256:512])
// One wave per row, shuffle reduction.
// ---------------------------------------------------------------------------
__global__ __launch_bounds__(256) void elr_kernel(
    const float* __restrict__ g, const float* __restrict__ attn_w,
    float* __restrict__ el, float* __restrict__ er)
{
    const int lane = threadIdx.x & 31;
    const int wv   = threadIdx.x >> 5;
    const int r    = blockIdx.x * 8 + wv;
    const float* grow = g + (size_t)r * DIM;

    float sl = 0.f, sr = 0.f;
    #pragma unroll
    for (int t = 0; t < 8; ++t) {
        const int f = lane + 32 * t;
        const float v = grow[f];
        sl += v * attn_w[f];
        sr += v * attn_w[DIM + f];
    }
    #pragma unroll
    for (int off = 16; off > 0; off >>= 1) {
        sl += __shfl_xor(sl, off, 32);
        sr += __shfl_xor(sr, off, 32);
    }
    if (lane == 0) { el[r] = sl; er[r] = sr; }
}

// ---------------------------------------------------------------------------
// Attention: out[i,h,:] = (sum_j w_ij * g[j,h,:]) / (sum_j w_ij),
// w_ij = exp(lrelu(el_i + er_j) - lrelu(el_i + max_j er_j)).
// Workgroup: 16 waves x 32 lanes, MI=32 i-rows, full 256 f-columns.
// Ping-pong LDS weight tile: iteration i generates tile i+1 into the alternate
// buffer, consumes tile i, then hits a single barrier. Denominators accumulate
// via ds_add_f32 with the same f16-rounded weights used in the numerator.
// ---------------------------------------------------------------------------
__global__ __launch_bounds__(512) void attn_kernel(
    const float* __restrict__ g, const float* __restrict__ el,
    const float* __restrict__ er, float* __restrict__ out)
{
    __shared__ _Float16 wtile[2][MI * 32];
    __shared__ float    den[MI];
    __shared__ float    red[512];

    const int tid  = threadIdx.x;
    const int h    = blockIdx.y;
    const int i0   = blockIdx.x * MI;
    const int lane = tid & 31;
    const int wv   = tid >> 5;
    const int hf   = lane >> 4;
    const int nc   = lane & 15;
    const int f0   = wv * 16;

    // per-head max of er (softmax stabilizer)
    float mx = -3.4e38f;
    for (int j = tid; j < N_TOK; j += 512)
        mx = fmaxf(mx, er[(size_t)j * HEADS + h]);
    red[tid] = mx;
    __syncthreads();
    #pragma unroll
    for (int s = 256; s > 0; s >>= 1) {
        if (tid < s) red[tid] = fmaxf(red[tid], red[tid + s]);
        __syncthreads();
    }
    const float ermax = red[0];

    if (tid < MI) den[tid] = 0.f;

    // tile-generator role: each thread owns (row, 2 adjacent j columns)
    const int   trow = tid >> 4;        // 0..31
    const int   tj0  = (tid & 15) * 2;  // 0..30
    const float elv  = el[(size_t)(i0 + trow) * HEADS + h];
    const float mrow = lrelu(elv + ermax);
    __syncthreads();   // den initialized, reduction done

    auto gen_tile = [&](int jbase, _Float16* wt) {
        float wsum = 0.f;
        #pragma unroll
        for (int q = 0; q < 2; ++q) {
            const int jj = tj0 + q;
            const float e = lrelu(elv + er[(size_t)(jbase + jj) * HEADS + h]);
            const _Float16 wh = (_Float16)__expf(e - mrow);
            wt[trow * 32 + jj] = wh;
            wsum += (float)wh;
        }
        atomicAdd(&den[trow], wsum);
    };

    const float* gcol = g + (size_t)h * DIM + f0 + nc;   // + j*HEADS*DIM

    v8f acc0 = {}, acc1 = {};

    gen_tile(0, wtile[0]);
    __syncthreads();

    for (int j0 = 0; j0 < N_TOK; j0 += 32) {
        const int buf = (j0 >> 5) & 1;
        const _Float16* wt = wtile[buf];

        // generate next tile into the alternate buffer (overlaps with consume)
        if (j0 + 32 < N_TOK) gen_tile(j0 + 32, wtile[buf ^ 1]);

        // B fragment: g[j0+K, h, f0+nc]; lane-half selects K 0..15 / 16..31
        v16h bfrag;
        #pragma unroll
        for (int t = 0; t < 16; ++t)
            bfrag[t] = (_Float16)gcol[(size_t)(j0 + 16 * hf + t) * (HEADS * DIM)];

        // A fragments from the current LDS weight tile
        v16h a0, a1;
        const int kb = 16 * hf;
        #pragma unroll
        for (int t = 0; t < 8; ++t) {
            a0[t]     = wt[nc * 32 + kb + t];
            a0[8 + t] = wt[nc * 32 + kb + 8 + t];
            a1[t]     = wt[(16 + nc) * 32 + kb + t];
            a1[8 + t] = wt[(16 + nc) * 32 + kb + 8 + t];
        }
        acc0 = __builtin_amdgcn_wmma_f32_16x16x32_f16(false, a0, false, bfrag,
                                                      (short)0, acc0, false, false);
        acc1 = __builtin_amdgcn_wmma_f32_16x16x32_f16(false, a1, false, bfrag,
                                                      (short)0, acc1, false, false);
        __syncthreads();   // single barrier per j-step (ping-pong spacing)
    }

    #pragma unroll
    for (int r = 0; r < 8; ++r) {
        const int m0r = r + 8 * hf;                 // 0..15
        const float inv0 = 1.f / den[m0r];
        const float inv1 = 1.f / den[16 + m0r];
        out[(size_t)((i0 + m0r) * HEADS + h) * DIM + f0 + nc]      = acc0[r] * inv0;
        out[(size_t)((i0 + 16 + m0r) * HEADS + h) * DIM + f0 + nc] = acc1[r] * inv1;
    }
}

// ---------------------------------------------------------------------------
extern "C" void kernel_launch(void* const* d_in, const int* in_sizes, int n_in,
                              void* d_out, int out_size, void* d_ws, size_t ws_size,
                              hipStream_t stream)
{
    const float* state  = (const float*)d_in[0];
    const float* left   = (const float*)d_in[1];
    /* d_in[2] = inputad (unused) */
    const float* W      = (const float*)d_in[3];
    const float* bias   = (const float*)d_in[4];
    const float* attn_w = (const float*)d_in[5];
    float* out = (float*)d_out;

    float* x  = (float*)d_ws;                 // ROWS*DIM f32 (16 MB)
    float* gm = x  + (size_t)ROWS * DIM;      // ROWS*DIM f32 (16 MB)
    float* el = gm + (size_t)ROWS * DIM;      // ROWS f32
    float* er = el + ROWS;                    // ROWS f32

    dim3 gblk(ROWS / 16, 2);                  // 1024 x 2 blocks, 8 waves each
    lin_kernel<<<gblk, 256, 0, stream>>>(0, left, state, nullptr, W, bias, x);
    lin_kernel<<<gblk, 256, 0, stream>>>(1, left, state, x,       W, bias, gm);
    elr_kernel<<<ROWS / 8, 256, 0, stream>>>(gm, attn_w, el, er);
    attn_kernel<<<dim3(N_TOK / MI, HEADS), 512, 0, stream>>>(gm, el, er, out);
}
